// Attention_188978561368
// MI455X (gfx1250) — matrix operand order repaired and checked by reading, not compiled, not run on hardware
//
#include <hip/hip_runtime.h>

typedef _Float16 h8   __attribute__((ext_vector_type(8)));
typedef _Float16 h16  __attribute__((ext_vector_type(16)));
typedef float    f8   __attribute__((ext_vector_type(8)));
typedef unsigned int u32x4 __attribute__((ext_vector_type(4)));
typedef int      i32x4 __attribute__((ext_vector_type(4)));
typedef int      i32x8 __attribute__((ext_vector_type(8)));

#define DIM    1024
#define T_SEQ  2048
#define B_SZ   4
#define HEADS  16
#define DHEAD  64
#define ROWS   (B_SZ * T_SEQ)   /* 8192 */
#define N_QKV  (3 * DIM)        /* 3072 */
#define SCALE  0.125f           /* 64^-0.5 */

#ifdef __has_builtin
#if __has_builtin(__builtin_amdgcn_tensor_load_to_lds)
#define ATT_HAVE_TDM 1
#endif
#endif
#ifndef ATT_HAVE_TDM
#define ATT_HAVE_TDM 0
#endif

__device__ __forceinline__ h16 cat8(h8 lo, h8 hi) {
  return __builtin_shufflevector(lo, hi, 0,1,2,3,4,5,6,7,8,9,10,11,12,13,14,15);
}

// A-fragment (16x32 f16) from a row-major matrix (row stride `stride` elems).
// ISA: lane L holds row L&15; K-chunks at kh*8 and 16+kh*8 (kh = L>>4).
__device__ __forceinline__ h16 frag_a(const _Float16* base, int stride, int lane) {
  const int row = lane & 15, kh = (lane >> 4) & 1;
  const _Float16* p = base + (size_t)row * stride + kh * 8;
  return cat8(*(const h8*)p, *(const h8*)(p + 16));
}

// B-fragment (32x16) from an N-major matrix Bt[n][k] (row stride `strideT`).
// ISA: lane L holds col L&15; K-chunks at kh*8 and 16+kh*8 (kh = L>>4).
__device__ __forceinline__ h16 frag_bt(const _Float16* baseT, int strideT, int lane) {
  const int n = lane & 15, kh = (lane >> 4) & 1;
  const _Float16* p = baseT + (size_t)n * strideT + kh * 8;
  return cat8(*(const h8*)p, *(const h8*)(p + 16));
}

__device__ __forceinline__ f8 wmma_f16(h16 a, h16 b, f8 c) {
  return __builtin_amdgcn_wmma_f32_16x16x32_f16(false, a, false, b, (short)0, c,
                                                false, false);
}

// Tensor Data Mover: 2D f16 tile (tile0 x tile1) global -> LDS, row stride
// `stride0` elements. Descriptor bit-packing per CDNA5 ISA ch. 8 (D# groups).
__device__ __forceinline__ void tdm_load_2d(void* ldsDst, const void* gSrc,
                                            unsigned tile0, unsigned tile1,
                                            unsigned tdim0, unsigned tdim1,
                                            unsigned long long stride0) {
#if ATT_HAVE_TDM
  const unsigned long long ga = (unsigned long long)(size_t)gSrc;
  u32x4 g0;
  g0[0] = 1u;                                        // count=1 (valid user D#)
  g0[1] = (unsigned)(size_t)ldsDst;                  // lds_addr (bytes)
  g0[2] = (unsigned)(ga & 0xFFFFFFFFu);              // global_addr[31:0]
  g0[3] = (unsigned)((ga >> 32) & 0x01FFFFFFu)       // global_addr[56:32]
        | (2u << 30);                                // type=2 ("image")
  i32x8 g1;
  g1[0] = (int)(1u << 16);                           // wg_mask=0, data_size=2B
  g1[1] = (int)((tdim0 & 0xFFFFu) << 16);            // tensor_dim0[15:0]
  g1[2] = (int)(((tdim0 >> 16) & 0xFFFFu) |          // tensor_dim0[31:16]
                ((tdim1 & 0xFFFFu) << 16));          // tensor_dim1[15:0]
  g1[3] = (int)(((tdim1 >> 16) & 0xFFFFu) |          // tensor_dim1[31:16]
                ((tile0 & 0xFFFFu) << 16));          // tile_dim0
  g1[4] = (int)(tile1 & 0xFFFFu);                    // tile_dim1 (tile_dim2=0)
  g1[5] = (int)(unsigned)(stride0 & 0xFFFFFFFFull);  // dim0_stride[31:0]
  g1[6] = (int)(unsigned)((stride0 >> 32) & 0xFFFFull); // dim0_stride[47:32]
  g1[7] = 0;                                         // dim1_stride (unused 2D)
  const i32x4 z4 = {0, 0, 0, 0};
#if __clang_major__ >= 23
  const i32x8 z8 = {0, 0, 0, 0, 0, 0, 0, 0};
  __builtin_amdgcn_tensor_load_to_lds(g0, g1, z4, z4, z8, 0);
#else
  __builtin_amdgcn_tensor_load_to_lds(g0, g1, z4, z4, 0);
#endif
#else
  (void)ldsDst; (void)gSrc; (void)tile0; (void)tile1;
  (void)tdim0; (void)tdim1; (void)stride0;
#endif
}

// ---------------------------------------------------------------------------
// Kernel 1: LayerNorm (f32 in) -> f16 normalized activations
// ---------------------------------------------------------------------------
__global__ __launch_bounds__(256) void ln_to_f16(const float* __restrict__ x,
                                                 const float* __restrict__ gamma,
                                                 const float* __restrict__ beta,
                                                 _Float16* __restrict__ xn) {
  const int row = blockIdx.x;
  const float* xr = x + (size_t)row * DIM;
  float s = 0.f, s2 = 0.f;
  for (int i = threadIdx.x; i < DIM; i += 256) {
    float v = xr[i];
    s += v; s2 += v * v;
  }
  for (int m = 16; m >= 1; m >>= 1) {
    s  += __shfl_xor(s,  m, 32);
    s2 += __shfl_xor(s2, m, 32);
  }
  __shared__ float rs[8], rs2[8], stats[2];
  const int wid = threadIdx.x >> 5, lane = threadIdx.x & 31;
  if (lane == 0) { rs[wid] = s; rs2[wid] = s2; }
  __syncthreads();
  if (threadIdx.x == 0) {
    float ts = 0.f, ts2 = 0.f;
    for (int i = 0; i < 8; i++) { ts += rs[i]; ts2 += rs2[i]; }
    const float mu = ts * (1.f / DIM);
    stats[0] = mu;
    stats[1] = rsqrtf(ts2 * (1.f / DIM) - mu * mu + 1e-5f);
  }
  __syncthreads();
  const float mu = stats[0], rstd = stats[1];
  for (int i = threadIdx.x; i < DIM; i += 256)
    xn[(size_t)row * DIM + i] = (_Float16)(((xr[i] - mu) * rstd) * gamma[i] + beta[i]);
}

// ---------------------------------------------------------------------------
// Kernel 2: weight f32 -> transposed f16:  in[K][N] f32 -> out[N][K] f16
// ---------------------------------------------------------------------------
__global__ __launch_bounds__(256) void cvt_t_f16(const float* __restrict__ in,
                                                 _Float16* __restrict__ out,
                                                 int K, int N) {
  const int idx = blockIdx.x * 256 + threadIdx.x;
  if (idx < K * N) {
    const int k = idx / N, n = idx % N;
    out[(size_t)n * K + k] = (_Float16)in[idx];
  }
}

// ---------------------------------------------------------------------------
// Kernel 3: LDS-free WMMA GEMM  C[M,N] = A[M,K] * Bt[N,K]^T
// (A row-major f16, B pre-transposed f16: fragments load straight from global,
//  everything L2-resident -> no LDS staging, no barriers.)
//   EPI=0: QKV epilogue -> Q,K row-major into qkv; V transposed into vt.
//   EPI=1: f32 output + bias.
// Block: 128 threads = 4 waves; wave computes 32x64; block tile 128x64.
// ---------------------------------------------------------------------------
template <int EPI>
__global__ __launch_bounds__(128) void gemm_wmma_k(const _Float16* __restrict__ A,
                                                   const _Float16* __restrict__ Bt,
                                                   void* __restrict__ Cv,
                                                   _Float16* __restrict__ aux,
                                                   const float* __restrict__ bias,
                                                   int M, int N, int K) {
  const int lane = threadIdx.x & 31, w = threadIdx.x >> 5;
  const int n0 = blockIdx.x * 64, m0 = blockIdx.y * 128 + w * 32;
  f8 acc[2][4] = {};

  for (int kk = 0; kk < K; kk += 32) {
    h16 af[2];
#pragma unroll
    for (int rt = 0; rt < 2; rt++)
      af[rt] = frag_a(A + (size_t)(m0 + rt * 16) * K + kk, K, lane);
    if (kk + 32 < K) {
      __builtin_prefetch(A + (size_t)m0 * K + kk + 32, 0, 1);
      __builtin_prefetch(Bt + (size_t)n0 * K + kk + 32, 0, 1);
    }
#pragma unroll
    for (int ct = 0; ct < 4; ct++) {
      const h16 bf = frag_bt(Bt + (size_t)(n0 + ct * 16) * K + kk, K, lane);
      acc[0][ct] = wmma_f16(af[0], bf, acc[0][ct]);
      acc[1][ct] = wmma_f16(af[1], bf, acc[1][ct]);
    }
  }

  // C/D layout: col = lane&15, rows = (lane>>4)*8 + i
  const int col = lane & 15, rb = (lane >> 4) << 3;
#pragma unroll
  for (int rt = 0; rt < 2; rt++) {
#pragma unroll
    for (int ct = 0; ct < 4; ct++) {
#pragma unroll
      for (int i = 0; i < 8; i++) {
        const int r = m0 + rt * 16 + rb + i;
        const int c = n0 + ct * 16 + col;
        if (EPI == 1) {
          ((float*)Cv)[(size_t)r * N + c] = acc[rt][ct][i] + bias[c];
        } else {
          const _Float16 hv = (_Float16)acc[rt][ct][i];
          if (c < 2 * DIM) {           // Q,K: row-major, head-packed
            ((_Float16*)Cv)[(size_t)r * N_QKV + c] = hv;
          } else {                     // V: transposed vt[bh][d][t]
            const int hd = c - 2 * DIM;
            const int b = r >> 11, t = r & (T_SEQ - 1);
            aux[(((size_t)b * HEADS + (hd >> 6)) * DHEAD + (hd & 63)) * T_SEQ + t] = hv;
          }
        }
      }
    }
  }
}

// ---------------------------------------------------------------------------
// Kernel 4: flash attention.
// K tile (row-major == N-major B layout) and V tile (pre-transposed vt) are
// transpose-free 2D tiles -> staged into LDS by the Tensor Data Mover.
// Grid: (T/64, B*H); block 128 = 4 waves; wave owns 16 query rows.
// ---------------------------------------------------------------------------
__global__ __launch_bounds__(128) void attn_fa(const _Float16* __restrict__ qkv,
                                               const _Float16* __restrict__ vtg,
                                               _Float16* __restrict__ aout) {
  const int bh = blockIdx.y, b = bh >> 4, h = bh & 15;
  const int lane = threadIdx.x & 31, w = threadIdx.x >> 5;
  const int q0 = blockIdx.x * 64 + w * 16;
  const _Float16* base  = qkv + (size_t)b * T_SEQ * N_QKV + h * DHEAD;
  const _Float16* kbase = base + DIM;
  const _Float16* vbase = vtg + (size_t)bh * DHEAD * T_SEQ;
  const int kh = (lane >> 4) & 1, n16 = lane & 15;

  // Q fragments straight from global in the ISA A-layout.
  h16 qa[2];
  {
    const _Float16* qp = base + (size_t)(q0 + n16) * N_QKV + kh * 8;
    qa[0] = cat8(*(const h8*)qp,        *(const h8*)(qp + 16));
    qa[1] = cat8(*(const h8*)(qp + 32), *(const h8*)(qp + 48));
  }

  __shared__ __align__(16) _Float16 Kt[64 * 64];      // K tile [kv][d]
  __shared__ __align__(16) _Float16 Vtl[64 * 64];     // V tile [d][kv]
  __shared__ __align__(16) _Float16 Pb[4][16 * 64];   // per-wave P tile
  _Float16* pb = Pb[w];

  f8 o[4] = {};
  float m[8], l[8];
#pragma unroll
  for (int i = 0; i < 8; i++) { m[i] = -INFINITY; l[i] = 0.f; }

  for (int kv0 = 0; kv0 < T_SEQ; kv0 += 64) {
    __syncthreads();            // all waves done reading previous tiles
#if ATT_HAVE_TDM
    if (w == 0) {
      tdm_load_2d(Kt,  kbase + (size_t)kv0 * N_QKV, 64, 64, 64, 64,
                  (unsigned long long)N_QKV);
      tdm_load_2d(Vtl, vbase + kv0, 64, 64, 64, 64,
                  (unsigned long long)T_SEQ);
      __builtin_amdgcn_s_wait_tensorcnt(0);
    }
#else
    {
      const int rr = threadIdx.x >> 1, hc = (threadIdx.x & 1) * 32;
      const _Float16* kp = kbase + (size_t)(kv0 + rr) * N_QKV + hc;
      const _Float16* vp = vbase + (size_t)rr * T_SEQ + kv0 + hc;
#pragma unroll
      for (int c = 0; c < 4; c++) {
        *(h8*)(Kt  + rr * 64 + hc + c * 8) = *(const h8*)(kp + c * 8);
        *(h8*)(Vtl + rr * 64 + hc + c * 8) = *(const h8*)(vp + c * 8);
      }
    }
#endif
    __syncthreads();

    // S = Q * K^T from LDS K tile.
    f8 s[4];
#pragma unroll
    for (int ct = 0; ct < 4; ct++) {
      f8 acc = {};
#pragma unroll
      for (int j = 0; j < 2; j++) {
        const h16 bf = frag_bt(Kt + ct * 16 * 64 + j * 32, 64, lane);
        acc = wmma_f16(qa[j], bf, acc);
      }
#pragma unroll
      for (int i = 0; i < 8; i++) acc[i] *= SCALE;
      s[ct] = acc;
    }

    // Online softmax; butterfly steps batched over all 8 rows so the
    // ds_bpermutes clause behind a single s_wait_dscnt.
    float mt[8];
#pragma unroll
    for (int i = 0; i < 8; i++)
      mt[i] = fmaxf(fmaxf(s[0][i], s[1][i]), fmaxf(s[2][i], s[3][i]));
#pragma unroll
    for (int mk = 8; mk >= 1; mk >>= 1) {
      float u[8];
#pragma unroll
      for (int i = 0; i < 8; i++) u[i] = __shfl_xor(mt[i], mk, 32);
#pragma unroll
      for (int i = 0; i < 8; i++) mt[i] = fmaxf(mt[i], u[i]);
    }
#pragma unroll
    for (int i = 0; i < 8; i++) {
      const float mn = fmaxf(m[i], mt[i]);
      const float cf = __expf(m[i] - mn);
      m[i] = mn; l[i] *= cf;
#pragma unroll
      for (int dt = 0; dt < 4; dt++) o[dt][i] *= cf;
    }
    float rsum[8];
#pragma unroll
    for (int i = 0; i < 8; i++) rsum[i] = 0.f;
#pragma unroll
    for (int ct = 0; ct < 4; ct++)
#pragma unroll
      for (int i = 0; i < 8; i++) {
        const float p = __expf(s[ct][i] - m[i]);
        s[ct][i] = p; rsum[i] += p;
      }
#pragma unroll
    for (int mk = 8; mk >= 1; mk >>= 1) {
      float u[8];
#pragma unroll
      for (int i = 0; i < 8; i++) u[i] = __shfl_xor(rsum[i], mk, 32);
#pragma unroll
      for (int i = 0; i < 8; i++) rsum[i] += u[i];
    }
#pragma unroll
    for (int i = 0; i < 8; i++) l[i] += rsum[i];

    // Reshape P (C-layout) -> A-layout via per-wave LDS tile (in-wave DS order).
    const int rb = (lane >> 4) * 8;
#pragma unroll
    for (int ct = 0; ct < 4; ct++)
#pragma unroll
      for (int i = 0; i < 8; i++)
        pb[(rb + i) * 64 + ct * 16 + n16] = (_Float16)s[ct][i];

    // O += P * V from LDS V tile.
#pragma unroll
    for (int j = 0; j < 2; j++) {
      const h16 pa = frag_a(pb + j * 32, 64, lane);
#pragma unroll
      for (int dt = 0; dt < 4; dt++) {
        const h16 vb = frag_bt(Vtl + dt * 16 * 64 + j * 32, 64, lane);
        o[dt] = wmma_f16(pa, vb, o[dt]);
      }
    }
  }

  // Epilogue: normalize, write head-concatenated f16 [B*T, 1024].
  const int rb = (lane >> 4) * 8;
#pragma unroll
  for (int i = 0; i < 8; i++) {
    const float inv = 1.f / l[i];
    const int r = q0 + rb + i;
#pragma unroll
    for (int dt = 0; dt < 4; dt++) {
      const int d = dt * 16 + n16;
      aout[((size_t)(b * T_SEQ + r)) * DIM + h * DHEAD + d] =
          (_Float16)(o[dt][i] * inv);
    }
  }
}

// ---------------------------------------------------------------------------
extern "C" void kernel_launch(void* const* d_in, const int* in_sizes, int n_in,
                              void* d_out, int out_size, void* d_ws, size_t ws_size,
                              hipStream_t stream) {
  (void)in_sizes; (void)n_in; (void)out_size; (void)ws_size;
  const float* x     = (const float*)d_in[0];
  const float* gamma = (const float*)d_in[1];
  const float* beta  = (const float*)d_in[2];
  const float* wqkv  = (const float*)d_in[3];
  const float* wout  = (const float*)d_in[4];
  const float* bout  = (const float*)d_in[5];
  float* out = (float*)d_out;

  // Workspace (all f16, ~110 MB total; L2-resident on 192 MB MI455X L2).
  _Float16* xn    = (_Float16*)d_ws;                        // 8192*1024
  _Float16* wqkvt = xn    + (size_t)ROWS * DIM;             // 3072*1024 (B^T)
  _Float16* woutt = wqkvt + (size_t)N_QKV * DIM;            // 1024*1024 (B^T)
  _Float16* qkv   = woutt + (size_t)DIM * DIM;              // 8192*3072 (Q,K)
  _Float16* vt    = qkv   + (size_t)ROWS * N_QKV;           // [64][64][2048]
  _Float16* attn  = vt    + (size_t)B_SZ * HEADS * DHEAD * T_SEQ; // 8192*1024

  ln_to_f16<<<ROWS, 256, 0, stream>>>(x, gamma, beta, xn);
  cvt_t_f16<<<(DIM * N_QKV + 255) / 256, 256, 0, stream>>>(wqkv, wqkvt, DIM, N_QKV);
  cvt_t_f16<<<(DIM * DIM + 255) / 256, 256, 0, stream>>>(wout, woutt, DIM, DIM);

  gemm_wmma_k<0><<<dim3(N_QKV / 64, ROWS / 128), 128, 0, stream>>>(
      xn, wqkvt, (void*)qkv, vt, nullptr, ROWS, N_QKV, DIM);

  attn_fa<<<dim3(T_SEQ / 64, B_SZ * HEADS), 128, 0, stream>>>(qkv, vt, attn);

  gemm_wmma_k<1><<<dim3(DIM / 64, ROWS / 128), 128, 0, stream>>>(
      attn, woutt, (void*)out, nullptr, bout, ROWS, DIM, DIM);
}
